// NetGIN_38671885533369
// MI455X (gfx1250) — matrix operand order repaired and checked by reading, not compiled, not run on hardware
//
#include <hip/hip_runtime.h>
#include <hip/hip_bf16.h>
#include <math.h>

#define N_NODES 50000
#define N_EDGES 800000
#define GDIM    64
#define N_LAYERS 5
#define TSTR    72   // padded LDS row stride (halves): 16B-aligned runs, bank-conflict-free

typedef __attribute__((ext_vector_type(16))) _Float16 v16h;
typedef __attribute__((ext_vector_type(8)))  _Float16 v8h;
typedef __attribute__((ext_vector_type(4)))  _Float16 v4h;
typedef __attribute__((ext_vector_type(8)))  float    v8f;

// A-matrix 16x32 f16 fragment per ISA 7.12.2: lane m=lane&15, khalf=lane>>4.
// half j in 0..7  -> K = kc*32 + 8*khalf + j        (contiguous 16B run)
// half j in 8..15 -> K = kc*32 + 8*khalf + 16 + (j-8) (second 16B run)
__device__ __forceinline__ v16h frag_a(const _Float16* tile, int lane, int kc) {
  const int m  = lane & 15;
  const int kh = lane >> 4;
  const _Float16* p = tile + m * TSTR + kc * 32 + kh * 8;
  v8h lo = *(const v8h*)(p);
  v8h hi = *(const v8h*)(p + 16);
  v16h a;
#pragma unroll
  for (int j = 0; j < 8; ++j) { a[j] = lo[j]; a[j + 8] = hi[j]; }
  return a;
}

// B-matrix 32x16 f16 fragment: lane n=(lane&15)+16*nt column, K = kc*32 + 8*(lane>>4) + j (j<8), +16 + (j-8) (j>=8).
// Weights stored transposed in LDS (wt[n*TSTR + k]) so the 16 K-halves are contiguous.
__device__ __forceinline__ v16h frag_b(const _Float16* wt, int lane, int kc, int nt) {
  const int n  = (lane & 15) + nt * 16;
  const int kh = lane >> 4;
  const _Float16* p = wt + n * TSTR + kc * 32 + kh * 8;
  v8h lo = *(const v8h*)(p);
  v8h hi = *(const v8h*)(p + 16);
  v16h b;
#pragma unroll
  for (int j = 0; j < 8; ++j) { b[j] = lo[j]; b[j + 8] = hi[j]; }
  return b;
}

// h = x, agg = x, pooled = 0
__global__ __launch_bounds__(256) void gin_init(const float* __restrict__ x,
    float* __restrict__ h, float* __restrict__ agg, float* __restrict__ pooled) {
  size_t tid = (size_t)blockIdx.x * 256 + threadIdx.x;          // one float4 each
  float4 v = ((const float4*)x)[tid];
  ((float4*)h)[tid]   = v;
  ((float4*)agg)[tid] = v;
  if (blockIdx.x == 0 && threadIdx.x < N_LAYERS * GDIM) pooled[threadIdx.x] = 0.0f;
}

// agg[dst] += h[src]  (16 lanes per edge, float4 gather, 4 scalar f32 atomics)
__global__ __launch_bounds__(256) void gin_scatter(const float* __restrict__ h,
    const int* __restrict__ src, const int* __restrict__ dst, float* __restrict__ agg) {
  unsigned tid = blockIdx.x * 256u + threadIdx.x;
  unsigned e = tid >> 4;
  int g = (int)(tid & 15u) << 2;
  int s = src[e], d = dst[e];
  float4 v = *(const float4*)(h + (size_t)s * GDIM + g);
  float* p = agg + (size_t)d * GDIM + g;
  atomicAdd(p + 0, v.x);
  atomicAdd(p + 1, v.y);
  atomicAdd(p + 2, v.z);
  atomicAdd(p + 3, v.w);
}

// Fused 3-stage MLP: relu(relu(relu(agg@W0)@W1)@W2) -> h, agg(=h), pooled col-sums.
// 8 waves/block, one 16-row node tile per wave, V_WMMA_F32_16X16X32_F16.
__global__ __launch_bounds__(256) void gin_mlp(
    const float* __restrict__ agg_in, const float* __restrict__ Wg, int layer,
    float* __restrict__ h_out, float* __restrict__ agg_out, float* __restrict__ pooled)
{
  __shared__ _Float16 sW[3][64 * TSTR];   // transposed f16 weights, 27.0 KB
  __shared__ _Float16 sT[8][16 * TSTR];   // per-wave f16 activation tiles, 18.0 KB

  const int tid  = threadIdx.x;
  const int wv   = tid >> 5;
  const int lane = tid & 31;
  const int tile = blockIdx.x * 8 + wv;
  const bool valid = tile < (N_NODES / 16);

  // fp32 -> fp16 weight load, transposed: sW[mi][n*TSTR + k] = W[layer][mi][k][n]
  const float* wbase = Wg + (size_t)layer * 3 * 64 * 64;
  for (int idx = tid; idx < 3 * 4096; idx += 256) {
    int mi  = idx >> 12;
    int rem = idx & 4095;
    int k   = rem >> 6;
    int n   = rem & 63;
    sW[mi][n * TSTR + k] = (_Float16)wbase[idx];
  }

  if (valid) {
    const float4* arow = (const float4*)(agg_in + (size_t)tile * 16 * GDIM);
#pragma unroll
    for (int it = 0; it < 8; ++it) {
      int idx = it * 32 + lane;              // 256 float4s = 16x64 tile
      int r   = idx >> 4;
      int c4  = (idx & 15) << 2;
      float4 v = arow[idx];
      v4h hv;
      hv[0] = (_Float16)v.x; hv[1] = (_Float16)v.y;
      hv[2] = (_Float16)v.z; hv[3] = (_Float16)v.w;
      *(v4h*)&sT[wv][r * TSTR + c4] = hv;
    }
  }
  __syncthreads();

  const int kh = lane >> 4;
  const int nl = lane & 15;

  // stages 0,1: GEMM + relu, output back into the per-wave LDS tile (C-layout -> rows)
#pragma unroll
  for (int s = 0; s < 2; ++s) {
    if (valid) {
      v16h a0 = frag_a(sT[wv], lane, 0);
      v16h a1 = frag_a(sT[wv], lane, 1);
#pragma unroll
      for (int nt = 0; nt < 4; ++nt) {
        v8f acc = {};
        v16h b0 = frag_b(sW[s], lane, 0, nt);
        v16h b1 = frag_b(sW[s], lane, 1, nt);
        acc = __builtin_amdgcn_wmma_f32_16x16x32_f16(false, a0, false, b0,
                                                     (short)0, acc, false, false);
        acc = __builtin_amdgcn_wmma_f32_16x16x32_f16(false, a1, false, b1,
                                                     (short)0, acc, false, false);
#pragma unroll
        for (int r = 0; r < 8; ++r) {
          float x = fmaxf(acc[r], 0.0f);
          int m = r + 8 * kh;                       // C layout: M = r + 8*(lane>>4)
          sT[wv][m * TSTR + nt * 16 + nl] = (_Float16)x;
        }
      }
    }
    __syncthreads();
  }

  // stage 2: GEMM + relu -> global h & agg(=h for next layer), pooled column sums
  if (valid) {
    v16h a0 = frag_a(sT[wv], lane, 0);
    v16h a1 = frag_a(sT[wv], lane, 1);
#pragma unroll
    for (int nt = 0; nt < 4; ++nt) {
      v8f acc = {};
      v16h b0 = frag_b(sW[2], lane, 0, nt);
      v16h b1 = frag_b(sW[2], lane, 1, nt);
      acc = __builtin_amdgcn_wmma_f32_16x16x32_f16(false, a0, false, b0,
                                                   (short)0, acc, false, false);
      acc = __builtin_amdgcn_wmma_f32_16x16x32_f16(false, a1, false, b1,
                                                   (short)0, acc, false, false);
      float colsum = 0.0f;
#pragma unroll
      for (int r = 0; r < 8; ++r) {
        float x = fmaxf(acc[r], 0.0f);
        int m = r + 8 * kh;
        size_t off = ((size_t)tile * 16 + m) * GDIM + nt * 16 + nl;
        h_out[off]   = x;
        agg_out[off] = x;
        colsum += x;
      }
      colsum += __shfl_xor(colsum, 16, 32);         // rows 0-7 half + rows 8-15 half
      if (lane < 16)
        atomicAdd(&pooled[layer * GDIM + nt * 16 + nl], colsum);
    }
  }
}

// logits = sum_l dot(pooled[l]/N, Wl[l]); out = sigmoid(sum)
__global__ __launch_bounds__(64) void gin_final(const float* __restrict__ pooled,
    const float* __restrict__ Wl, float* __restrict__ out) {
  __shared__ float red[64];
  int d = threadIdx.x;
  float acc = 0.0f;
#pragma unroll
  for (int l = 0; l < N_LAYERS; ++l) acc += pooled[l * GDIM + d] * Wl[l * GDIM + d];
  red[d] = acc * (1.0f / (float)N_NODES);
  __syncthreads();
  if (d == 0) {
    float s = 0.0f;
#pragma unroll
    for (int i = 0; i < 64; ++i) s += red[i];
    out[0] = 1.0f / (1.0f + expf(-s));
  }
}

extern "C" void kernel_launch(void* const* d_in, const int* in_sizes, int n_in,
                              void* d_out, int out_size, void* d_ws, size_t ws_size,
                              hipStream_t stream) {
  const float* x   = (const float*)d_in[0];   // [50000,64]
  const float* W   = (const float*)d_in[1];   // [5,3,64,64]
  const float* Wl  = (const float*)d_in[2];   // [5,64]
  const int*   src = (const int*)d_in[3];     // [800000]
  const int*   dst = (const int*)d_in[4];     // [800000]
  float* out = (float*)d_out;

  float* h      = (float*)d_ws;                         // 12.8 MB
  float* agg    = h   + (size_t)N_NODES * GDIM;         // 12.8 MB
  float* pooled = agg + (size_t)N_NODES * GDIM;         // 1.25 KB

  // 50000*64/4 float4s / 256 = 3125 blocks (exact)
  gin_init<<<3125, 256, 0, stream>>>(x, h, agg, pooled);

  for (int l = 0; l < N_LAYERS; ++l) {
    // 800000 edges * 16 lanes / 256 = 50000 blocks (exact)
    gin_scatter<<<50000, 256, 0, stream>>>(h, src, dst, agg);
    // 3125 tiles, 8 per block -> 391 blocks
    gin_mlp<<<391, 256, 0, stream>>>(agg, W, l, h, agg, pooled);
  }

  gin_final<<<1, 64, 0, stream>>>(pooled, Wl, out);
}
